// RelBlock_81071802679764
// MI455X (gfx1250) — compile-verified, matrix operand order checked by reading, not run on hardware
//
#include <hip/hip_runtime.h>

typedef _Float16 h16;
typedef unsigned int u32;
typedef __attribute__((ext_vector_type(16))) _Float16 v16h;
typedef __attribute__((ext_vector_type(8)))  float    v8f;

#define BD  4
#define TD  1024
#define CD  768
#define HH  12
#define HDD 64

union FragH { v16h v; uint4 q[2]; h16 s[16]; };
union Frag8 { uint4 q; h16 s[8]; };

__device__ __forceinline__ v16h ldg_frag(const h16* p) {
  FragH u;
  u.q[0] = *(const uint4*)p;
  u.q[1] = *(const uint4*)(p + 16);
  return u.v;
}

__device__ __forceinline__ v16h lds_frag(const h16* base, int ldt, int lane) {
  const h16* p = base + (lane & 15) * ldt + ((lane >> 4) * 8);
  FragH u;
  u.q[0] = *(const uint4*)p;
  u.q[1] = *(const uint4*)(p + 16);
  return u.v;
}

__device__ __forceinline__ v8f wmma_f16(v16h a, v16h b, v8f c) {
  return __builtin_amdgcn_wmma_f32_16x16x32_f16(false, a, false, b, (short)0, c,
                                                false, false);
}

// ---------------- weight convert (f32 [K][N] -> f16 transposed [N][K]) -----
__global__ void convt_k(const float* __restrict__ w, h16* __restrict__ wt,
                        int K, int N) {
  size_t i = (size_t)blockIdx.x * 256 + threadIdx.x;
  if (i >= (size_t)K * N) return;
  int k = (int)(i / N), n = (int)(i % N);
  wt[(size_t)n * K + k] = (h16)w[i];
}

// ---------------- sinusoidal relative-position table [1024][64] ------------
__global__ void rpe_k(h16* __restrict__ tab) {
  int i = blockIdx.x * 256 + threadIdx.x;
  if (i >= 1024 * 64) return;
  int d = i >> 6, c = i & 63;
  int j = c & 31;
  float invf = powf(10000.f, -(float)(2 * j) / 64.f);
  float f = (float)d * invf;
  tab[i] = (h16)((c < 32) ? sinf(f) : cosf(f));
}

// ---------------- layernorm (row of 768) -> f16 ----------------------------
__global__ __launch_bounds__(256)
void ln_k(const float* __restrict__ x, const float* __restrict__ w,
          h16* __restrict__ out) {
  __shared__ float red[256];
  const int row = blockIdx.x, tid = threadIdx.x;
  const float* xr = x + (size_t)row * CD;
  float s = 0.f;
  for (int c = tid; c < CD; c += 256) s += xr[c];
  red[tid] = s; __syncthreads();
  for (int st = 128; st > 0; st >>= 1) {
    if (tid < st) red[tid] += red[tid + st];
    __syncthreads();
  }
  float mean = red[0] / (float)CD;
  __syncthreads();
  float v = 0.f;
  for (int c = tid; c < CD; c += 256) { float d = xr[c] - mean; v += d * d; }
  red[tid] = v; __syncthreads();
  for (int st = 128; st > 0; st >>= 1) {
    if (tid < st) red[tid] += red[tid + st];
    __syncthreads();
  }
  float rstd = rsqrtf(red[0] / (float)CD + 1e-5f);
  for (int c = tid; c < CD; c += 256)
    out[(size_t)row * CD + c] = (h16)((xr[c] - mean) * rstd * w[c]);
}

// ---------------- generic f16 WMMA GEMM: C = A[M,K] @ BT[N,K]^T ------------
// EPI 0: store f16   1: out = resid + acc (f32)   2: gelu -> f16   3: out += acc
template<int EPI>
__global__ __launch_bounds__(256)
void gemm_k(const h16* __restrict__ A, const h16* __restrict__ BT,
            int M, int N, int K,
            float* __restrict__ outF, h16* __restrict__ outB,
            const float* __restrict__ resid) {
  constexpr int LDT = 48;                       // 32 + 16 pad (16B aligned rows)
  __shared__ __align__(16) h16 As[64 * LDT];
  __shared__ __align__(16) h16 Bs[128 * LDT];
  const int m0 = blockIdx.x * 64;
  const int n0 = blockIdx.y * 128;
  const int tid = threadIdx.x, lane = tid & 31, wave = tid >> 5;
  const int wm = (wave >> 2) * 32;
  const int wn = (wave & 3) * 32;
  v8f acc[2][2] = {};
  const int arow = tid >> 2, aseg = (tid & 3) * 8;
  const int brow = tid >> 1, bseg = (tid & 1) * 16;

  for (int k0 = 0; k0 < K; k0 += 32) {
    *(uint4*)&As[arow * LDT + aseg] =
        *(const uint4*)&A[(size_t)(m0 + arow) * K + k0 + aseg];
    *(uint4*)&Bs[brow * LDT + bseg] =
        *(const uint4*)&BT[(size_t)(n0 + brow) * K + k0 + bseg];
    *(uint4*)&Bs[brow * LDT + bseg + 8] =
        *(const uint4*)&BT[(size_t)(n0 + brow) * K + k0 + bseg + 8];
    __syncthreads();
    v16h a0 = lds_frag(&As[(wm + 0) * LDT], LDT, lane);
    v16h a1 = lds_frag(&As[(wm + 16) * LDT], LDT, lane);
    v16h b0 = lds_frag(&Bs[(wn + 0) * LDT], LDT, lane);
    v16h b1 = lds_frag(&Bs[(wn + 16) * LDT], LDT, lane);
    acc[0][0] = wmma_f16(a0, b0, acc[0][0]);
    acc[0][1] = wmma_f16(a0, b1, acc[0][1]);
    acc[1][0] = wmma_f16(a1, b0, acc[1][0]);
    acc[1][1] = wmma_f16(a1, b1, acc[1][1]);
    __syncthreads();
  }

  const int mr = (lane >> 4) * 8;
  const int nc = lane & 15;
#pragma unroll
  for (int ti = 0; ti < 2; ++ti)
#pragma unroll
    for (int tj = 0; tj < 2; ++tj)
#pragma unroll
      for (int v = 0; v < 8; ++v) {
        int gm = m0 + wm + ti * 16 + mr + v;
        int gn = n0 + wn + tj * 16 + nc;
        size_t idx = (size_t)gm * N + gn;
        float val = acc[ti][tj][v];
        if constexpr (EPI == 0) {
          outB[idx] = (h16)val;
        } else if constexpr (EPI == 1) {
          outF[idx] = resid[idx] + val;
        } else if constexpr (EPI == 2) {
          float g = 0.5f * val * (1.0f + erff(val * 0.70710678118654752f));
          outB[idx] = (h16)g;
        } else {
          outF[idx] += val;
        }
      }
}

// ---------------- V transpose: qkv v-chunk -> vT[b,h,hd,T] (LDS tiled) -----
__global__ __launch_bounds__(256)
void vtrans_k(const h16* __restrict__ qkv, h16* __restrict__ vT) {
  __shared__ __align__(16) h16 tile[64 * 72];   // 72 = 64 + 8 pad (16B rows)
  const int t0 = blockIdx.x * 64;
  const int bh = blockIdx.y;
  const int b = bh / HH, h = bh % HH;
  const int tid = threadIdx.x;
  {
    int tt = tid >> 2, seg = (tid & 3) * 16;
    const h16* src =
        qkv + (size_t)(b * TD + t0 + tt) * (3 * CD) + 2 * CD + h * HDD + seg;
    *(uint4*)&tile[tt * 72 + seg]     = *(const uint4*)src;
    *(uint4*)&tile[tt * 72 + seg + 8] = *(const uint4*)(src + 8);
  }
  __syncthreads();
  {
    int d = tid >> 2, seg = (tid & 3) * 16;
    Frag8 u0, u1;
#pragma unroll
    for (int e = 0; e < 8; ++e) {
      u0.s[e] = tile[(seg + e) * 72 + d];
      u1.s[e] = tile[(seg + 8 + e) * 72 + d];
    }
    h16* dst = vT + ((size_t)bh * HDD + d) * TD + t0 + seg;
    *(uint4*)dst       = u0.q;
    *(uint4*)(dst + 8) = u1.q;
  }
}

// ---------------- qtab = q @ table^T, rows r = (b*T+t)*H + h ---------------
__global__ __launch_bounds__(256)
void qtab_k(const h16* __restrict__ qkv, const h16* __restrict__ tab,
            h16* __restrict__ qtab) {
  const int r0 = blockIdx.x * 128;
  const int n0 = blockIdx.y * 64;
  {  // causal trim: distances above the block's max token index are never read
    int tFirst = r0 / HH, tLast = (r0 + 127) / HH;
    if ((tFirst >> 10) == (tLast >> 10) && n0 > (tLast & 1023)) return;
  }
  const int tid = threadIdx.x, lane = tid & 31, wave = tid >> 5;
  const int rw = r0 + wave * 16;
  const int lr = lane & 15, koff = (lane >> 4) * 8;
  const int rr = rw + lr;
  const int tok = rr / HH, h = rr % HH;
  const h16* ab = qkv + (size_t)tok * (3 * CD) + h * HDD;
  v16h a0 = ldg_frag(ab + koff);
  v16h a1 = ldg_frag(ab + 32 + koff);
  const int mr = (lane >> 4) * 8;
#pragma unroll
  for (int j = 0; j < 4; ++j) {
    const h16* bb = tab + (size_t)(n0 + j * 16 + lr) * HDD;
    v16h b0 = ldg_frag(bb + koff);
    v16h b1 = ldg_frag(bb + 32 + koff);
    v8f c = {};
    c = wmma_f16(a0, b0, c);
    c = wmma_f16(a1, b1, c);
#pragma unroll
    for (int v = 0; v < 8; ++v)
      qtab[(size_t)(rw + mr + v) * 1024 + n0 + j * 16 + lr] = (h16)c[v];
  }
}

// ---------------- causal flash attention with relative-position gather -----
__global__ __launch_bounds__(128)
void flash_k(const h16* __restrict__ qkv, const h16* __restrict__ vT,
             const h16* __restrict__ qtab, h16* __restrict__ yout) {
  __shared__ __align__(16) h16 Pst[4][16 * 48];  // per-wave P tile 16x32
  const int bh = blockIdx.y;
  const int b = bh / HH, h = bh % HH;
  const int tid = threadIdx.x, lane = tid & 31, wave = tid >> 5;
  const int i0 = blockIdx.x * 64 + wave * 16;
  const int lr = lane & 15, hi = lane >> 4;
  const int koff = hi * 8;

  const h16* qb = qkv + (size_t)(b * TD + i0 + lr) * (3 * CD) + h * HDD;
  v16h qa0 = ldg_frag(qb + koff);
  v16h qa1 = ldg_frag(qb + 32 + koff);

  float mrow[8], lrow[8];
  v8f acc[4] = {};
#pragma unroll
  for (int v = 0; v < 8; ++v) { mrow[v] = -1e30f; lrow[v] = 0.f; }

  h16* P = Pst[wave];
  const int jmax = i0 + 15;                       // j0 <= 992, spans <= 1023
  for (int j0 = 0; j0 <= jmax; j0 += 32) {
    if (j0 + 32 <= jmax) {                        // L2 -> WGP prefetch
      __builtin_prefetch(
          qkv + (size_t)(b * TD + j0 + 32 + lr) * (3 * CD) + CD + h * HDD, 0, 0);
      __builtin_prefetch(vT + ((size_t)bh * HDD + lr) * TD + j0 + 32, 0, 0);
    }
    v8f s[2];
#pragma unroll
    for (int tj = 0; tj < 2; ++tj) {       // S = Q @ K^T  (two 16-col tiles)
      int jn = j0 + tj * 16 + lr;
      const h16* kb = qkv + (size_t)(b * TD + jn) * (3 * CD) + CD + h * HDD;
      v16h kb0 = ldg_frag(kb + koff);
      v16h kb1 = ldg_frag(kb + 32 + koff);
      v8f z = {};
      z = wmma_f16(qa0, kb0, z);
      z = wmma_f16(qa1, kb1, z);
      s[tj] = z;
    }
#pragma unroll
    for (int tj = 0; tj < 2; ++tj)         // + rel gather, scale, causal mask
#pragma unroll
      for (int v = 0; v < 8; ++v) {
        int i = i0 + hi * 8 + v;
        int j = j0 + tj * 16 + lr;
        float sv = -1e30f;
        if (j <= i) {
          float rel =
              (float)qtab[((size_t)(b * TD + i) * HH + h) * 1024 + (i - j)];
          sv = (s[tj][v] + rel) * 0.125f;
        }
        s[tj][v] = sv;
      }
    float alpha[8];
#pragma unroll
    for (int v = 0; v < 8; ++v) {          // online-softmax max update
      float mv = fmaxf(s[0][v], s[1][v]);
      mv = fmaxf(mv, __shfl_xor(mv, 1, 32));
      mv = fmaxf(mv, __shfl_xor(mv, 2, 32));
      mv = fmaxf(mv, __shfl_xor(mv, 4, 32));
      mv = fmaxf(mv, __shfl_xor(mv, 8, 32));
      float mnew = fmaxf(mrow[v], mv);
      alpha[v] = __expf(mrow[v] - mnew);
      mrow[v] = mnew;
      lrow[v] *= alpha[v];
    }
#pragma unroll
    for (int t = 0; t < 4; ++t)
#pragma unroll
      for (int v = 0; v < 8; ++v) acc[t][v] *= alpha[v];
#pragma unroll
    for (int v = 0; v < 8; ++v) {          // p = exp(s-m), row-sum, stage P
      float p0 = (s[0][v] <= -1e29f) ? 0.f : __expf(s[0][v] - mrow[v]);
      float p1 = (s[1][v] <= -1e29f) ? 0.f : __expf(s[1][v] - mrow[v]);
      float rs = p0 + p1;
      rs += __shfl_xor(rs, 1, 32);
      rs += __shfl_xor(rs, 2, 32);
      rs += __shfl_xor(rs, 4, 32);
      rs += __shfl_xor(rs, 8, 32);
      lrow[v] += rs;
      int m = hi * 8 + v;
      P[m * 48 + lr] = (h16)p0;
      P[m * 48 + 16 + lr] = (h16)p1;
    }
    asm volatile("s_wait_dscnt 0" ::: "memory");  // C-layout -> A-layout via LDS
    v16h pa = lds_frag(P, 48, lane);
#pragma unroll
    for (int th = 0; th < 4; ++th) {       // acc += P @ V (vT is K-contiguous)
      const h16* vb = vT + ((size_t)bh * HDD + th * 16 + lr) * TD + j0;
      acc[th] = wmma_f16(pa, ldg_frag(vb + koff), acc[th]);
    }
  }
#pragma unroll
  for (int th = 0; th < 4; ++th)
#pragma unroll
    for (int v = 0; v < 8; ++v) {
      int i = i0 + hi * 8 + v;
      float y = acc[th][v] / lrow[v];
      yout[(size_t)(b * TD + i) * CD + h * HDD + th * 16 + lr] = (h16)y;
    }
}

extern "C" void kernel_launch(void* const* d_in, const int* in_sizes, int n_in,
                              void* d_out, int out_size, void* d_ws,
                              size_t ws_size, hipStream_t stream) {
  const float* x      = (const float*)d_in[0];
  const float* ln1w   = (const float*)d_in[1];
  const float* wattn  = (const float*)d_in[2];
  // d_in[3] = w_pos: dead code in the reference, never used
  const float* wcproj = (const float*)d_in[4];
  const float* ln2w   = (const float*)d_in[5];
  const float* wfc    = (const float*)d_in[6];
  const float* wmproj = (const float*)d_in[7];
  float* out = (float*)d_out;
  (void)in_sizes; (void)n_in; (void)out_size; (void)ws_size;

  h16* ws = (h16*)d_ws;
  size_t o = 0;
  auto take = [&](size_t n) {
    h16* p = ws + o;
    o += (n + 127) & ~(size_t)127;
    return p;
  };
  h16* wAt   = take((size_t)2304 * 768);    // w_attn^T
  h16* wCt   = take((size_t)768 * 768);     // w_cproj^T
  h16* wFt   = take((size_t)3072 * 768);    // w_fc^T
  h16* wMt   = take((size_t)768 * 3072);    // w_mproj^T
  h16* hB    = take((size_t)4096 * 768);    // LN output (reused for LN2)
  h16* qkvB  = take((size_t)4096 * 2304);   // fused qkv
  h16* tabB  = take((size_t)1024 * 64);     // rpe table (d >= 0)
  h16* qtabB = take((size_t)49152 * 1024);  // q @ table^T
  h16* vTB   = take((size_t)48 * 64 * 1024 + 128);  // V transposed per (b,h)
  h16* yB    = take((size_t)4096 * 768);    // attention output
  h16* fcB   = take((size_t)4096 * 3072);   // gelu(fc) output

  convt_k<<<(768 * 2304 + 255) / 256, 256, 0, stream>>>(wattn, wAt, 768, 2304);
  convt_k<<<(768 * 768 + 255) / 256, 256, 0, stream>>>(wcproj, wCt, 768, 768);
  convt_k<<<(768 * 3072 + 255) / 256, 256, 0, stream>>>(wfc, wFt, 768, 3072);
  convt_k<<<(3072 * 768 + 255) / 256, 256, 0, stream>>>(wmproj, wMt, 3072, 768);
  rpe_k<<<(1024 * 64) / 256, 256, 0, stream>>>(tabB);

  ln_k<<<4096, 256, 0, stream>>>(x, ln1w, hB);
  gemm_k<0><<<dim3(64, 18), 256, 0, stream>>>(hB, wAt, 4096, 2304, 768,
                                              nullptr, qkvB, nullptr);
  vtrans_k<<<dim3(16, 48), 256, 0, stream>>>(qkvB, vTB);
  qtab_k<<<dim3(384, 16), 256, 0, stream>>>(qkvB, tabB, qtabB);
  flash_k<<<dim3(16, 48), 128, 0, stream>>>(qkvB, vTB, qtabB, yB);
  gemm_k<1><<<dim3(64, 6), 256, 0, stream>>>(yB, wCt, 4096, 768, 768, out,
                                             nullptr, x);
  ln_k<<<4096, 256, 0, stream>>>(out, ln2w, hB);
  gemm_k<2><<<dim3(64, 24), 256, 0, stream>>>(hB, wFt, 4096, 3072, 768,
                                              nullptr, fcB, nullptr);
  gemm_k<3><<<dim3(64, 6), 256, 0, stream>>>(fcB, wMt, 4096, 768, 3072, out,
                                             nullptr, nullptr);
}
